// LSTM_43061342110215
// MI455X (gfx1250) — matrix operand order repaired and checked by reading, not compiled, not run on hardware
//
#include <hip/hip_runtime.h>
#include <math.h>
#include <stdint.h>

// ---------------------------------------------------------------------------
// LSTM on MI455X (gfx1250): bf16 WMMA recurrence, TDM-staged weights in LDS,
// f32 cell state.   B=128, T=256, D=256, H=512, C=128
// ---------------------------------------------------------------------------

typedef __bf16 bf16_t;
typedef __attribute__((ext_vector_type(16))) __bf16 v16bf;
typedef __attribute__((ext_vector_type(8)))  __bf16 v8bf;
typedef __attribute__((ext_vector_type(8)))  float  v8f;
typedef __attribute__((ext_vector_type(4)))  unsigned int u32x4;
typedef __attribute__((ext_vector_type(8)))  int i32x8;
typedef __attribute__((ext_vector_type(4)))  int i32x4;

#define BB 128
#define TT 256
#define DD 256
#define HH 512
#define CC 128
#define KCAT 768           // H + D
#define NGATE 2048         // 4 * H
#define NSTRIP 32          // N columns staged per block via TDM

// ---- workspace layout (bytes) ---------------------------------------------
#define OFF_WCAT  0                 // bf16 [2048 x 768]   3,145,728
#define OFF_WPHB  3145728           // bf16 [128 x 512]      131,072
#define OFF_XT    3276800           // bf16 [T x B x D]   16,777,216
#define OFF_H     20054016          // bf16 [B x H]          131,072
#define OFF_C     20185088          // f32  [B x H]          262,144
#define OFF_GA    20447232          // f32  [B x 2048]     1,048,576

// ---------------------------------------------------------------------------
// A 16x32 bf16 fragment: lane L -> row M = L&15; half hb = L>>4;
//   elems e hold K = k0 + hb*16 + e
__device__ __forceinline__ v16bf load_fragA(const bf16_t* __restrict__ base,
                                            int row0, int k0, int ld, int lane) {
  const int r  = lane & 15;
  const int hb = lane >> 4;
  const bf16_t* p = base + (size_t)(row0 + r) * ld + (k0 + hb * 16);
  return *reinterpret_cast<const v16bf*>(p);
}

// B 32x16 bf16 with B[k][n] = W[n][k], W row-major (N x K):
// lane L -> col N = n0 + (L&15); elems e hold K = k0 + (L>>4)*16 + e
__device__ __forceinline__ v16bf load_fragB_g(const bf16_t* __restrict__ W,
                                              int n0, int k0, int ld, int lane) {
  const bf16_t* p = W + (size_t)(n0 + (lane & 15)) * ld + (k0 + (lane >> 4) * 16);
  return *reinterpret_cast<const v16bf*>(p);
}

__device__ __forceinline__ v16bf load_fragB_lds(const bf16_t* sB,
                                                int n0, int k0, int lane) {
  const bf16_t* p = sB + (size_t)(n0 + (lane & 15)) * KCAT + (k0 + (lane >> 4) * 16);
  return *reinterpret_cast<const v16bf*>(p);
}

// ---------------------------------------------------------------------------
// TDM: stage a 2D row-major bf16 tile [tile_rows x KCAT] of Wcat into LDS.
// D# per ISA ch.8: group0 {count=1, lds_addr, global_addr, type=2},
// group1 {data_size=2B, tensor 768 x 2048, tile 768 x tile_rows, stride 768}.
__device__ __forceinline__ void tdm_load_wcat_strip(const bf16_t* gsrc,
                                                    unsigned lds_off, int tile_rows) {
  unsigned long long ga = (unsigned long long)(uintptr_t)gsrc;
  u32x4 g0;
  g0[0] = 1u;                                    // count=1, user descriptor
  g0[1] = lds_off;                               // lds_addr (bytes, from real ptr)
  g0[2] = (unsigned)(ga & 0xFFFFFFFFu);          // global_addr[31:0]
  g0[3] = (unsigned)(ga >> 32) | (2u << 30);     // global_addr[56:32] | type=2
  i32x8 g1;
  g1[0] = 0x10000;                               // workgroup_mask=0, data_size=1 (2B)
  g1[1] = (int)((KCAT & 0xFFFF) << 16);          // tensor_dim0 lo16 @ [31:16]
  g1[2] = (int)(((unsigned)KCAT >> 16) | ((NGATE & 0xFFFF) << 16)); // dim0 hi | dim1 lo
  g1[3] = (int)(((unsigned)NGATE >> 16) | ((KCAT & 0xFFFF) << 16)); // dim1 hi | tile_dim0
  g1[4] = tile_rows;                             // tile_dim1 lo16; tile_dim2 = 0
  g1[5] = KCAT;                                  // tensor_dim0_stride lo32
  g1[6] = 0;                                     // stride hi | tensor_dim1_stride lo
  g1[7] = 0;
  i32x4 gz = {0, 0, 0, 0};
#if defined(__clang_major__) && __clang_major__ >= 23
  i32x8 gz8 = {0, 0, 0, 0, 0, 0, 0, 0};
  __builtin_amdgcn_tensor_load_to_lds(g0, g1, gz, gz, gz8, 0);
#else
  __builtin_amdgcn_tensor_load_to_lds(g0, g1, gz, gz, 0);
#endif
}

// ---------------------------------------------------------------------------
// Prep: Wcat (2048 x 768 bf16) = [Wh | Wx] per gate {g,i,f,o}; cast Wph
__global__ void prep_weights(const float* __restrict__ Wgx, const float* __restrict__ Wgh,
                             const float* __restrict__ Wix, const float* __restrict__ Wih,
                             const float* __restrict__ Wfx, const float* __restrict__ Wfh,
                             const float* __restrict__ Wox, const float* __restrict__ Woh,
                             const float* __restrict__ Wph,
                             bf16_t* __restrict__ Wcat, bf16_t* __restrict__ WphB) {
  const int NCAT = NGATE * KCAT;
  int idx = blockIdx.x * blockDim.x + threadIdx.x;
  if (idx < NCAT) {
    int n = idx / KCAT, k = idx - n * KCAT;
    int g = n >> 9, r = n & 511;
    const float* Wh[4] = {Wgh, Wih, Wfh, Woh};
    const float* Wx[4] = {Wgx, Wix, Wfx, Wox};
    float v = (k < HH) ? Wh[g][(size_t)r * HH + k]
                       : Wx[g][(size_t)r * DD + (k - HH)];
    Wcat[idx] = (bf16_t)v;
  } else {
    int i2 = idx - NCAT;
    if (i2 < CC * HH) WphB[i2] = (bf16_t)Wph[i2];
  }
}

// Gather embeddings -> xt bf16, layout (t, b, d)
__global__ void embed_gather(const long long* __restrict__ x,   // (B, T, 1) int64
                             const float* __restrict__ emb,     // (C, D)
                             bf16_t* __restrict__ xt) {
  int idx = blockIdx.x * blockDim.x + threadIdx.x;   // T*B*D
  int d  = idx & (DD - 1);
  int tb = idx >> 8;
  int b  = tb & (BB - 1);
  int t  = tb >> 7;
  long long tok = x[(size_t)b * TT + t];
  xt[idx] = (bf16_t)emb[(size_t)tok * DD + d];
}

__global__ void init_state(float* __restrict__ c, bf16_t* __restrict__ h) {
  int idx = blockIdx.x * blockDim.x + threadIdx.x;   // B*H
  c[idx] = 0.0f;
  h[idx] = (bf16_t)0.0f;
}

// ---------------------------------------------------------------------------
// One recurrence step: ga = [h_{t-1} | x_t] (128x768) @ Wcat^T (768x2048)
// grid = 128 blocks x 128 threads (4 waves).
// Block -> 64 M-rows x 32 N-cols; its Wcat strip (32x768 bf16, 48KB) staged
// into LDS by one TDM descriptor, consumed as WMMA B-fragments via ds loads.
__global__ void lstm_step_gemm(const bf16_t* __restrict__ Wcat,
                               const bf16_t* __restrict__ xt,
                               const bf16_t* __restrict__ hprev,
                               float* __restrict__ ga,
                               int t) {
  __shared__ __align__(32) bf16_t sB[NSTRIP * KCAT];   // 48 KB
  const int lane  = threadIdx.x & 31;
  const int wv    = threadIdx.x >> 5;           // 0..3
  const int mblk  = blockIdx.x & 1;             // M half
  const int nstr  = blockIdx.x >> 1;            // 0..63
  const int row0  = mblk * 64 + wv * 16;
  const int col0  = nstr * NSTRIP;
  const bf16_t* xbase = xt + (size_t)t * BB * DD;

  if (wv == 0) {
    // lds offset from the real pointer (addr[31:0] of a generic LDS pointer
    // is the byte offset inside the workgroup allocation) -- this also makes
    // sB escape so the LDS reads below cannot be folded away.
    unsigned lds_off = (unsigned)(size_t)(void*)&sB[0];
    tdm_load_wcat_strip(Wcat + (size_t)col0 * KCAT, lds_off, NSTRIP);
    __builtin_amdgcn_s_wait_tensorcnt(0);
  }
  __syncthreads();

  // Launder the LDS base so alias analysis treats it as written by the TDM.
  bf16_t* sBr = sB;
  asm volatile("" : "+v"(sBr) : : "memory");

  v8f acc[2] = {};

  // K = 0..511 from h_{t-1}
  for (int k0 = 0; k0 < HH; k0 += 32) {
    v16bf a = load_fragA(hprev, row0, k0, HH, lane);
#pragma unroll
    for (int j = 0; j < 2; ++j) {
      v16bf b = load_fragB_lds(sBr, j * 16, k0, lane);
      acc[j] = __builtin_amdgcn_wmma_f32_16x16x32_bf16(false, a, false, b,
                                                       (short)0, acc[j], false, false);
    }
  }
  // K = 512..767 from x_t
  for (int k0 = HH; k0 < KCAT; k0 += 32) {
    v16bf a = load_fragA(xbase, row0, k0 - HH, DD, lane);
#pragma unroll
    for (int j = 0; j < 2; ++j) {
      v16bf b = load_fragB_lds(sBr, j * 16, k0, lane);
      acc[j] = __builtin_amdgcn_wmma_f32_16x16x32_bf16(false, a, false, b,
                                                       (short)0, acc[j], false, false);
    }
  }

  const int cn = lane & 15;
  const int cm = (lane >> 4) * 8;
#pragma unroll
  for (int j = 0; j < 2; ++j)
#pragma unroll
    for (int v = 0; v < 8; ++v)
      ga[(size_t)(row0 + cm + v) * NGATE + (col0 + j * 16 + cn)] = acc[j][v];
}

// Elementwise cell update: bias + nonlinearities, c f32, h -> bf16
__global__ void lstm_cell(const float* __restrict__ ga,
                          const float* __restrict__ bg, const float* __restrict__ bi,
                          const float* __restrict__ bfb, const float* __restrict__ bo,
                          float* __restrict__ c, bf16_t* __restrict__ h) {
  int idx = blockIdx.x * blockDim.x + threadIdx.x;   // B*H
  int b = idx >> 9, j = idx & 511;
  const float* row = ga + (size_t)b * NGATE;
  float g = tanhf(row[j] + bg[j]);
  float i = 1.0f / (1.0f + expf(-(row[HH + j]     + bi[j])));
  float f = 1.0f / (1.0f + expf(-(row[2 * HH + j] + bfb[j])));
  float o = 1.0f / (1.0f + expf(-(row[3 * HH + j] + bo[j])));
  float cv = g * i + c[idx] * f;
  c[idx] = cv;
  h[idx] = (bf16_t)(tanhf(cv) * o);
}

// ---------------------------------------------------------------------------
// Head: logits = h @ Wph^T + bp ; out = log_softmax(logits, axis=1)
__global__ void logits_logsoftmax(const bf16_t* __restrict__ h,
                                  const bf16_t* __restrict__ WphB,
                                  const float* __restrict__ bp,
                                  float* __restrict__ out) {
  __shared__ float sl[BB * CC];     // 64 KB
  const int lane = threadIdx.x & 31;
  const int wid  = threadIdx.x >> 5;      // 0..7
  const int row0 = wid * 16;

  v8f acc[8] = {};
  for (int k0 = 0; k0 < HH; k0 += 32) {
    v16bf a = load_fragA(h, row0, k0, HH, lane);
#pragma unroll
    for (int n = 0; n < 8; ++n) {
      v16bf b = load_fragB_g(WphB, n * 16, k0, HH, lane);
      acc[n] = __builtin_amdgcn_wmma_f32_16x16x32_bf16(false, a, false, b,
                                                       (short)0, acc[n], false, false);
    }
  }
  const int cn = lane & 15;
  const int cm = (lane >> 4) * 8;
#pragma unroll
  for (int n = 0; n < 8; ++n)
#pragma unroll
    for (int v = 0; v < 8; ++v)
      sl[(row0 + cm + v) * CC + (n * 16 + cn)] = acc[n][v] + bp[n * 16 + cn];

  __syncthreads();

  if (threadIdx.x < BB) {
    const int r = threadIdx.x;
    float mx = -INFINITY;
    for (int cI = 0; cI < CC; ++cI) mx = fmaxf(mx, sl[r * CC + cI]);
    float s = 0.0f;
    for (int cI = 0; cI < CC; ++cI) s += expf(sl[r * CC + cI] - mx);
    float lse = mx + logf(s);
    for (int cI = 0; cI < CC; ++cI) out[r * CC + cI] = sl[r * CC + cI] - lse;
  }
}

// ---------------------------------------------------------------------------
extern "C" void kernel_launch(void* const* d_in, const int* in_sizes, int n_in,
                              void* d_out, int out_size, void* d_ws, size_t ws_size,
                              hipStream_t stream) {
  const long long* x  = (const long long*)d_in[0];
  const float* emb    = (const float*)d_in[1];
  const float* Wgx    = (const float*)d_in[2];
  const float* Wgh    = (const float*)d_in[3];
  const float* bg     = (const float*)d_in[4];
  const float* Wix    = (const float*)d_in[5];
  const float* Wih    = (const float*)d_in[6];
  const float* bi     = (const float*)d_in[7];
  const float* Wfx    = (const float*)d_in[8];
  const float* Wfh    = (const float*)d_in[9];
  const float* bfb    = (const float*)d_in[10];
  const float* Wox    = (const float*)d_in[11];
  const float* Woh    = (const float*)d_in[12];
  const float* bo     = (const float*)d_in[13];
  const float* Wph    = (const float*)d_in[14];
  const float* bp     = (const float*)d_in[15];
  float* out          = (float*)d_out;

  char* ws = (char*)d_ws;
  bf16_t* Wcat = (bf16_t*)(ws + OFF_WCAT);
  bf16_t* WphB = (bf16_t*)(ws + OFF_WPHB);
  bf16_t* xt   = (bf16_t*)(ws + OFF_XT);
  bf16_t* hB   = (bf16_t*)(ws + OFF_H);
  float*  cS   = (float*)(ws + OFF_C);
  float*  ga   = (float*)(ws + OFF_GA);

  prep_weights<<<(NGATE * KCAT + CC * HH) / 256, 256, 0, stream>>>(
      Wgx, Wgh, Wix, Wih, Wfx, Wfh, Wox, Woh, Wph, Wcat, WphB);
  embed_gather<<<(TT * BB * DD) / 256, 256, 0, stream>>>(x, emb, xt);
  init_state<<<(BB * HH) / 256, 256, 0, stream>>>(cS, hB);

  for (int t = 0; t < TT; ++t) {
    lstm_step_gemm<<<128, 128, 0, stream>>>(Wcat, xt, hB, ga, t);
    lstm_cell<<<(BB * HH) / 256, 256, 0, stream>>>(ga, bg, bi, bfb, bo, cS, hB);
  }

  logits_logsoftmax<<<1, 256, 0, stream>>>(hB, WphB, bp, out);
}
